// FlashMHA_50929722196717
// MI455X (gfx1250) — matrix operand order repaired and checked by reading, not compiled, not run on hardware
//
#include <hip/hip_runtime.h>

typedef __attribute__((ext_vector_type(16))) __bf16 v16bf;
typedef __attribute__((ext_vector_type(16))) short  v16s;
typedef __attribute__((ext_vector_type(8)))  short  v8s;
typedef __attribute__((ext_vector_type(8)))  float  v8f;

#define BB 4
#define TT 2048
#define CC 768
#define HH 12
#define DD 64
#define BT (BB*TT)   /* 8192 */
#define C3 (3*CC)    /* 2304 */

// ---------- helpers ----------

__device__ __forceinline__ unsigned short f2bf(float f) {
  unsigned u = __builtin_bit_cast(unsigned, f);
  u += 0x7FFFu + ((u >> 16) & 1u);      // round-to-nearest-even
  return (unsigned short)(u >> 16);
}

// A-fragment, 16x32 bf16 (row-major source, reduction dim contiguous).
// lanes 0-15: M=0..15; elems 0..7 -> K=k0+8g+0..7, elems 8..15 -> K=k0+16+8g+0..7
__device__ __forceinline__ v16bf load_a_frag(const unsigned short* base,
                                             int ld, int row0, int k0, int lane) {
  int m = row0 + (lane & 15);
  int g = lane >> 4;
  const unsigned short* p = base + (size_t)m * ld + k0 + g * 8;
  v8s lo = *(const v8s*)p;
  v8s hi = *(const v8s*)(p + 16);
  v16s s = __builtin_shufflevector(lo, hi, 0,1,2,3,4,5,6,7,8,9,10,11,12,13,14,15);
  return __builtin_bit_cast(v16bf, s);
}

// B-fragment, 32x16 bf16 from "transposed" storage: base[n*ld + k].
// lane n = col0 + (lane&15); elems 0..7 -> K=k0+8g+0..7, elems 8..15 -> K=k0+16+8g+0..7
__device__ __forceinline__ v16bf load_b_frag(const unsigned short* base,
                                             int ld, int col0, int k0, int lane) {
  int n = col0 + (lane & 15);
  int g = lane >> 4;
  const unsigned short* p = base + (size_t)n * ld + k0 + g * 8;
  v8s lo = *(const v8s*)p;
  v8s hi = *(const v8s*)(p + 16);
  v16s s = __builtin_shufflevector(lo, hi, 0,1,2,3,4,5,6,7,8,9,10,11,12,13,14,15);
  return __builtin_bit_cast(v16bf, s);
}

__device__ __forceinline__ v8f wmma_bf16(v16bf a, v16bf b, v8f c) {
  return __builtin_amdgcn_wmma_f32_16x16x32_bf16(false, a, false, b, (short)0, c,
                                                 false, false);
}

// Async DMA: 16 bytes global -> LDS, tracked by ASYNCcnt.
// LDS generic addresses carry the LDS byte offset in the low 32 bits (flat aperture rule).
__device__ __forceinline__ void async_b128(const unsigned short* g, unsigned short* l) {
  unsigned loff = (unsigned)(unsigned long long)l;
  asm volatile("global_load_async_to_lds_b128 %0, %1, off"
               :: "v"(loff), "v"((unsigned long long)g) : "memory");
}
__device__ __forceinline__ void wait_async0() {
  asm volatile("s_wait_asynccnt 0x0" ::: "memory");
}

// ---------- conversion kernels ----------

__global__ void k_cvt(const float* __restrict__ src, unsigned short* __restrict__ dst, int n) {
  int i = blockIdx.x * blockDim.x + threadIdx.x;
  if (i < n) dst[i] = f2bf(src[i]);
}

// src [rows, cols] fp32 -> dst [cols, rows] bf16
__global__ void k_cvt_t(const float* __restrict__ src, unsigned short* __restrict__ dst,
                        int rows, int cols) {
  int i = blockIdx.x * blockDim.x + threadIdx.x;
  if (i < rows * cols) {
    int r = i / cols, c = i % cols;
    dst[(size_t)c * rows + r] = f2bf(src[i]);
  }
}

// ---------- fused QKV projection: qkv = x @ Wqkv + bqkv ----------
// Each wave computes a 32x64 tile: 2 A-frags + 4 B-frags -> 8 WMMAs per k-step.
// Writes Q (scaled by 1/sqrt(D)) as [BH,T,D], K as [BH,T,D], V transposed as [BH,D,T].
__global__ __launch_bounds__(128) void k_qkv(const unsigned short* __restrict__ xb,
                                             const unsigned short* __restrict__ wt,
                                             const float* __restrict__ bqkv,
                                             unsigned short* __restrict__ Qb,
                                             unsigned short* __restrict__ Kb,
                                             unsigned short* __restrict__ Vt) {
  int lane = threadIdx.x & 31;
  int wave = threadIdx.x >> 5;
  int row0 = blockIdx.x * 32;
  int col0 = (blockIdx.y * 4 + wave) * 64;   // 36 col-groups over 3C

  v8f acc[2][4] = {};
#pragma unroll 2
  for (int k0 = 0; k0 < CC; k0 += 32) {
    v16bf a0 = load_a_frag(xb, CC, row0,      k0, lane);
    v16bf a1 = load_a_frag(xb, CC, row0 + 16, k0, lane);
#pragma unroll
    for (int j = 0; j < 4; ++j) {
      v16bf b = load_b_frag(wt, CC, col0 + 16 * j, k0, lane);
      acc[0][j] = wmma_bf16(a0, b, acc[0][j]);
      acc[1][j] = wmma_bf16(a1, b, acc[1][j]);
    }
  }

  int g = lane >> 4, ln = lane & 15;
#pragma unroll
  for (int j = 0; j < 4; ++j) {
    int n = col0 + 16 * j + ln;
    float bias = bqkv[n];
    int which = n / CC;        // 0=Q 1=K 2=V
    int c = n % CC;
    int h = c / DD, d = c % DD;
#pragma unroll
    for (int i = 0; i < 2; ++i) {
#pragma unroll
      for (int r = 0; r < 8; ++r) {
        int row = row0 + 16 * i + r + 8 * g;
        int bidx = row / TT, t = row % TT;
        float v = acc[i][j][r] + bias;
        size_t bh = (size_t)(bidx * HH + h);
        if (which == 0)      Qb[(bh * TT + t) * DD + d] = f2bf(v * 0.125f); // fold 1/sqrt(64)
        else if (which == 1) Kb[(bh * TT + t) * DD + d] = f2bf(v);
        else                 Vt[(bh * DD + d) * TT + t] = f2bf(v);
      }
    }
  }
}

// ---------- flash attention ----------
// Block = 4 waves, one (b,h), 128 queries. K/V chunks (32 keys) are DMA'd into LDS
// once per block with GLOBAL_LOAD_ASYNC_TO_LDS_B128 (double-buffered, ASYNCcnt),
// then all 4 waves read fragments from LDS: 16 WMMAs per chunk per wave.
__global__ __launch_bounds__(128) void k_attn(const unsigned short* __restrict__ Qb,
                                              const unsigned short* __restrict__ Kb,
                                              const unsigned short* __restrict__ Vt,
                                              const long long* __restrict__ amask,
                                              unsigned short* __restrict__ ctx) {
  __shared__ __attribute__((aligned(64))) unsigned short sK[2][32 * DD]; // [key][d]
  __shared__ __attribute__((aligned(64))) unsigned short sV[2][DD * 32]; // [d][key]
  __shared__ __attribute__((aligned(32))) unsigned short sP[4][2][16 * 32];

  int tid  = threadIdx.x;
  int lane = tid & 31;
  int wave = tid >> 5;
  int g = lane >> 4, ln = lane & 15;
  int bh = blockIdx.y;                   // 0..BH-1
  int b  = bh / HH, h = bh % HH;
  int qrow0 = (blockIdx.x * 4 + wave) * 32;

  const unsigned short* Qbase = Qb + (size_t)bh * TT * DD;
  const unsigned short* Kbase = Kb + (size_t)bh * TT * DD;
  const unsigned short* Vbase = Vt + (size_t)bh * DD * TT;
  const long long* mrow = amask + (size_t)b * TT;

  // cooperative stage of one 32-key chunk: K 4KB (contiguous) + V 4KB (64 rows x 64B)
  auto stage = [&](int buf, int kt) {
    const unsigned short* gk = Kbase + (size_t)kt * DD + tid * 16;
    unsigned short* lk = &sK[buf][tid * 16];
    async_b128(gk, lk);
    async_b128(gk + 8, lk + 8);
    int row = tid >> 1, half = tid & 1;
    const unsigned short* gv = Vbase + (size_t)row * TT + kt + half * 16;
    unsigned short* lv = &sV[buf][row * 32 + half * 16];
    async_b128(gv, lv);
    async_b128(gv + 8, lv + 8);
  };

  v16bf qa[2][2];
#pragma unroll
  for (int t = 0; t < 2; ++t) {
    qa[t][0] = load_a_frag(Qbase, DD, qrow0 + 16 * t, 0,  lane);
    qa[t][1] = load_a_frag(Qbase, DD, qrow0 + 16 * t, 32, lane);
  }

  float m[2][8], l[2][8];
  v8f o[2][4] = {};
#pragma unroll
  for (int t = 0; t < 2; ++t)
#pragma unroll
    for (int r = 0; r < 8; ++r) { m[t][r] = -__builtin_inff(); l[t][r] = 0.f; }

  const float FMIN = -3.402823466e38f;   // finfo(f32).min, matches reference mask fill

  stage(0, 0);
  for (int kt = 0; kt < TT; kt += 32) {
    int cur = (kt >> 5) & 1;
    wait_async0();        // own DMA portion done
    __syncthreads();      // everyone's portion done; previous buffer now dead
    if (kt + 32 < TT) stage(cur ^ 1, kt + 32);   // overlap next DMA with compute

    const unsigned short* ldsK = sK[cur];
    const unsigned short* ldsV = sV[cur];

    // K fragments from LDS, shared by both query sub-tiles: kb[key-half][d-half]
    v16bf kb[2][2];
    kb[0][0] = load_b_frag(ldsK, DD, 0,  0,  lane);
    kb[0][1] = load_b_frag(ldsK, DD, 0,  32, lane);
    kb[1][0] = load_b_frag(ldsK, DD, 16, 0,  lane);
    kb[1][1] = load_b_frag(ldsK, DD, 16, 32, lane);

    // S tiles: s[t][key-half] (scale pre-folded into Q)
    v8f s[2][2] = {};
#pragma unroll
    for (int t = 0; t < 2; ++t)
#pragma unroll
      for (int kh = 0; kh < 2; ++kh) {
        s[t][kh] = wmma_bf16(qa[t][0], kb[kh][0], s[t][kh]);
        s[t][kh] = wmma_bf16(qa[t][1], kb[kh][1], s[t][kh]);
      }

    bool keep0 = (mrow[kt + ln] == 0);        // reference quirk: attend where mask==0
    bool keep1 = (mrow[kt + 16 + ln] == 0);

#pragma unroll
    for (int t = 0; t < 2; ++t) {
      unsigned short* pt = sP[wave][t];
#pragma unroll
      for (int r = 0; r < 8; ++r) {
        float a0 = keep0 ? s[t][0][r] : FMIN;
        float a1 = keep1 ? s[t][1][r] : FMIN;
        // row max across the 16 lanes holding this row's columns
        float mx = fmaxf(a0, a1);
        mx = fmaxf(mx, __shfl_xor(mx, 1, 32));
        mx = fmaxf(mx, __shfl_xor(mx, 2, 32));
        mx = fmaxf(mx, __shfl_xor(mx, 4, 32));
        mx = fmaxf(mx, __shfl_xor(mx, 8, 32));
        float nm = fmaxf(m[t][r], mx);
        float alpha = __expf(m[t][r] - nm);
        float p0 = __expf(a0 - nm);
        float p1 = __expf(a1 - nm);
        float ps = p0 + p1;
        ps += __shfl_xor(ps, 1, 32);
        ps += __shfl_xor(ps, 2, 32);
        ps += __shfl_xor(ps, 4, 32);
        ps += __shfl_xor(ps, 8, 32);
        m[t][r] = nm;
        l[t][r] = l[t][r] * alpha + ps;
        o[t][0][r] *= alpha; o[t][1][r] *= alpha;
        o[t][2][r] *= alpha; o[t][3][r] *= alpha;
        // stage P (C-layout -> LDS row-major) for re-load as A fragment
        int row = r + 8 * g;
        pt[row * 32 + ln]      = f2bf(p0);
        pt[row * 32 + 16 + ln] = f2bf(p1);
      }
    }
    asm volatile("s_wait_dscnt 0x0" ::: "memory");

    // P as A fragments (16x32 each) from LDS
    v16bf pa[2];
#pragma unroll
    for (int t = 0; t < 2; ++t) {
      const unsigned short* pp = sP[wave][t] + (size_t)ln * 32 + g * 8;
      v8s lo = *(const v8s*)pp;
      v8s hi = *(const v8s*)(pp + 16);
      v16s ps16 = __builtin_shufflevector(lo, hi, 0,1,2,3,4,5,6,7,8,9,10,11,12,13,14,15);
      pa[t] = __builtin_bit_cast(v16bf, ps16);
    }

    // O(32x64) += P(32x32) @ V(32x64), V fragments from LDS (shared)
#pragma unroll
    for (int j = 0; j < 4; ++j) {
      v16bf vb = load_b_frag(ldsV, 32, 16 * j, 0, lane);
      o[0][j] = wmma_bf16(pa[0], vb, o[0][j]);
      o[1][j] = wmma_bf16(pa[1], vb, o[1][j]);
    }
  }

  // normalize and emit context in [B,T,C] bf16 layout for the output GEMM
  unsigned short* cbase = ctx + (size_t)b * TT * CC + h * DD;
#pragma unroll
  for (int t = 0; t < 2; ++t) {
#pragma unroll
    for (int r = 0; r < 8; ++r) {
      float inv = 1.0f / l[t][r];
      int tok = qrow0 + 16 * t + r + 8 * g;
      unsigned short* crow = cbase + (size_t)tok * CC;
      crow[ln]      = f2bf(o[t][0][r] * inv);
      crow[16 + ln] = f2bf(o[t][1][r] * inv);
      crow[32 + ln] = f2bf(o[t][2][r] * inv);
      crow[48 + ln] = f2bf(o[t][3][r] * inv);
    }
  }
}

// ---------- output projection: out = ctx @ Wo + bo (fp32 out) ----------
// 32x64 tile per wave, same blocking as k_qkv.
__global__ __launch_bounds__(128) void k_oproj(const unsigned short* __restrict__ ctx,
                                               const unsigned short* __restrict__ wot,
                                               const float* __restrict__ bo,
                                               float* __restrict__ out) {
  int lane = threadIdx.x & 31;
  int wave = threadIdx.x >> 5;
  int row0 = blockIdx.x * 32;
  int col0 = (blockIdx.y * 4 + wave) * 64;   // 12 col-groups

  v8f acc[2][4] = {};
#pragma unroll 2
  for (int k0 = 0; k0 < CC; k0 += 32) {
    v16bf a0 = load_a_frag(ctx, CC, row0,      k0, lane);
    v16bf a1 = load_a_frag(ctx, CC, row0 + 16, k0, lane);
#pragma unroll
    for (int j = 0; j < 4; ++j) {
      v16bf b = load_b_frag(wot, CC, col0 + 16 * j, k0, lane);
      acc[0][j] = wmma_bf16(a0, b, acc[0][j]);
      acc[1][j] = wmma_bf16(a1, b, acc[1][j]);
    }
  }
  int g = lane >> 4, ln = lane & 15;
#pragma unroll
  for (int j = 0; j < 4; ++j) {
    float bias = bo[col0 + 16 * j + ln];
#pragma unroll
    for (int i = 0; i < 2; ++i)
#pragma unroll
      for (int r = 0; r < 8; ++r)
        out[(size_t)(row0 + 16 * i + r + 8 * g) * CC + col0 + 16 * j + ln] =
            acc[i][j][r] + bias;
  }
}

// ---------- launcher ----------

extern "C" void kernel_launch(void* const* d_in, const int* in_sizes, int n_in,
                              void* d_out, int out_size, void* d_ws, size_t ws_size,
                              hipStream_t stream) {
  (void)in_sizes; (void)n_in; (void)out_size; (void)ws_size;
  const float*     x     = (const float*)d_in[0];
  const long long* amask = (const long long*)d_in[1];
  const float*     Wqkv  = (const float*)d_in[2];
  const float*     bqkv  = (const float*)d_in[3];
  const float*     Wo    = (const float*)d_in[4];
  const float*     bo    = (const float*)d_in[5];

  char* ws = (char*)d_ws;
  size_t off = 0;
  auto alloc = [&](size_t nelem) {
    unsigned short* p = (unsigned short*)(ws + off);
    off += nelem * sizeof(unsigned short);
    off = (off + 255) & ~(size_t)255;
    return p;
  };
  unsigned short* xb    = alloc((size_t)BT * CC);            // x in bf16
  unsigned short* wqkvT = alloc((size_t)C3 * CC);            // Wqkv^T bf16
  unsigned short* woT   = alloc((size_t)CC * CC);            // Wo^T bf16
  unsigned short* Qb    = alloc((size_t)BB * HH * TT * DD);  // [BH,T,D] (pre-scaled)
  unsigned short* Kb    = alloc((size_t)BB * HH * TT * DD);  // [BH,T,D]
  unsigned short* Vt    = alloc((size_t)BB * HH * DD * TT);  // [BH,D,T]
  unsigned short* ctx   = alloc((size_t)BT * CC);            // [B,T,C]

  int nx = BT * CC;
  k_cvt  <<<(nx + 255) / 256, 256, 0, stream>>>(x, xb, nx);
  k_cvt_t<<<(CC * C3 + 255) / 256, 256, 0, stream>>>(Wqkv, wqkvT, CC, C3);
  k_cvt_t<<<(CC * CC + 255) / 256, 256, 0, stream>>>(Wo, woT, CC, CC);

  k_qkv  <<<dim3(BT / 32, C3 / 64 / 4), 128, 0, stream>>>(xb, wqkvT, bqkv, Qb, Kb, Vt);
  k_attn <<<dim3(TT / 32 / 4, BB * HH), 128, 0, stream>>>(Qb, Kb, Vt, amask, ctx);
  k_oproj<<<dim3(BT / 32, CC / 64 / 4), 128, 0, stream>>>(ctx, woT, bo, (float*)d_out);
}
